// VectorQuantizer_4389456576699
// MI455X (gfx1250) — compile-verified, matrix operand order checked by reading, not run on hardware
//
#include <hip/hip_runtime.h>
#include <hip/hip_bf16.h>

// ---------------------------------------------------------------------------
// Problem constants
// ---------------------------------------------------------------------------
#define NB      4096
#define NSUB    16
#define ED      64
#define NE      1024
#define NF      65536          // NB*NSUB rows of zf
#define NBINS   256
#define INV_T   (1.0f/0.07f)

typedef unsigned long long u64;
typedef unsigned int       u32;
typedef unsigned short     u16;

// WMMA / TDM vector types (CDNA5, wave32)
typedef __attribute__((ext_vector_type(16))) __bf16       v16bf;
typedef __attribute__((ext_vector_type(8)))  float        v8f;
typedef __attribute__((ext_vector_type(4)))  unsigned int u32x4;
typedef __attribute__((ext_vector_type(8)))  int          i32x8;
typedef __attribute__((ext_vector_type(4)))  int          i32x4;

// Tensor Data Mover availability (arity differs across toolchains)
#if defined(__has_builtin)
#  if __has_builtin(__builtin_amdgcn_tensor_load_to_lds) && \
      __has_builtin(__builtin_amdgcn_s_wait_tensorcnt)
#    define USE_TDM 1
#  endif
#endif
#ifndef USE_TDM
#  define USE_TDM 0
#endif

// ---------------------------------------------------------------------------
// Output layout (floats), tuple concatenated flat:
// (loss, loss1, loss2, z_q_st, z_output, perplexity, min_encodings, idx,
//  new_embed_prob, new_emb)
// ---------------------------------------------------------------------------
#define O_LOSS   0
#define O_LOSS1  1
#define O_LOSS2  2
#define O_ZQST   3
#define O_ZOUT   (3 + 4194304)
#define O_PERP   (3 + 2*4194304)          // 8388611
#define O_ME     (4 + 2*4194304)          // 8388612, 67108864 floats == d buffer
#define O_IDX    (O_ME + 67108864)        // 75497476
#define O_NEP    (O_IDX + 65536)          // 75563012
#define O_NEMB   (O_NEP + 1024)           // 75564036

// ---------------------------------------------------------------------------
// Workspace layout (bytes, 256-aligned)
// ---------------------------------------------------------------------------
#define WS_ZN       0u                     // NF*ED bf16 = 8 MB
#define WS_CN       8388608u               // NE*ED bf16 = 128 KB
#define WS_ROWIDX   8519680u               // NF int
#define WS_COLPACK  8781824u               // NE u64
#define WS_HIST     8790016u               // NE*NBINS int = 1 MB
#define WS_THRTOP   9838592u               // NE float
#define WS_THRMED   9842688u               // NE float
#define WS_TOPSUM   9846784u               // NE float
#define WS_TOPCNT   9850880u               // NE int
#define WS_SUMEXP   9854976u               // NE float
#define WS_ECNT     9859072u               // NE int
#define WS_SCAL     9863168u               // scalars: [0]=sq_sum [1]=contra [2]=entropy
#define WS_END      9863424u

// ---------------------------------------------------------------------------
// Helpers
// ---------------------------------------------------------------------------
__device__ __forceinline__ u32 fkey(float f) {
    // monotonic float -> uint mapping (for atomicMax-based argmax)
    u32 u = __float_as_uint(f);
    return (u & 0x80000000u) ? ~u : (u | 0x80000000u);
}
__device__ __forceinline__ u16 f2bf(float f) {
    // round-to-nearest-even fp32 -> bf16
    u32 u = __float_as_uint(f);
    u32 r = u + 0x7FFFu + ((u >> 16) & 1u);
    return (u16)(r >> 16);
}

#if USE_TDM
// Issue one TDM tile load: `rows` rows x 64 bf16 (= 16 units of 8B) from a
// row-major bf16 matrix into LDS at byte offset `lds_off`.
// D# packing per CDNA5 ISA (count=1, type=2 "image").
__device__ __forceinline__ void tdm_load_tile(unsigned lds_off,
                                              const u16* __restrict__ gbase,
                                              unsigned rows, unsigned total_rows) {
    unsigned long long ga = (unsigned long long)(uintptr_t)gbase;
    u32x4 g0;
    g0[0] = 1u;                                              // count=1, user mode
    g0[1] = lds_off;                                         // lds_addr (bytes)
    g0[2] = (unsigned)(ga & 0xFFFFFFFFull);                  // global_addr[31:0]
    g0[3] = (unsigned)((ga >> 32) & 0x01FFFFFFull)           // global_addr[56:32]
          | (2u << 30);                                      // type=2
    i32x8 g1;
    g1[0] = (int)(3u << 16);                                 // data_size=3 (8B), mask=0
    g1[1] = (int)(16u << 16);                                // tensor_dim0[15:0]=16
    g1[2] = (int)((total_rows & 0xFFFFu) << 16);             // tensor_dim1[15:0]
    g1[3] = (int)((total_rows >> 16) | (16u << 16));         // tensor_dim1[31:16] | tile_dim0=16
    g1[4] = (int)rows;                                       // tile_dim1 (tile_dim2=0)
    g1[5] = 16;                                              // tensor_dim0_stride=16
    g1[6] = 0;
    g1[7] = 0;
    i32x4 gz = {0, 0, 0, 0};
#if __clang_major__ >= 23
    i32x8 gz8 = {0, 0, 0, 0, 0, 0, 0, 0};
    __builtin_amdgcn_tensor_load_to_lds(g0, g1, gz, gz, gz8, 0);
#else
    __builtin_amdgcn_tensor_load_to_lds(g0, g1, gz, gz, 0);
#endif
}
#endif

// ---------------------------------------------------------------------------
// 1) Row L2-normalize fp32[rows][64] -> bf16[rows][64]; one wave32 per row.
// ---------------------------------------------------------------------------
__global__ __launch_bounds__(256) void k_norm_rows(const float* __restrict__ in,
                                                   u16* __restrict__ outbf,
                                                   int nrows) {
    int row  = blockIdx.x * 8 + (threadIdx.x >> 5);
    int lane = threadIdx.x & 31;
    if (row >= nrows) return;
    float v0 = in[(size_t)row * ED + lane * 2 + 0];
    float v1 = in[(size_t)row * ED + lane * 2 + 1];
    float s  = v0 * v0 + v1 * v1;
    #pragma unroll
    for (int off = 16; off > 0; off >>= 1) s += __shfl_xor(s, off, 32);
    float inv = 1.0f / fmaxf(sqrtf(s), 1e-12f);
    outbf[(size_t)row * ED + lane * 2 + 0] = f2bf(v0 * inv);
    outbf[(size_t)row * ED + lane * 2 + 1] = f2bf(v1 * inv);
}

// ---------------------------------------------------------------------------
// 2) WMMA GEMM: d[NF,NE] = zn(bf16) @ cn(bf16)^T, fp32 out.
//    Block tile 128(M) x 64(N), full K=64 staged in LDS (TDM if available).
//    8 waves/block; each wave owns 16 M-rows, sweeps 4 N-subtiles reusing its
//    A fragments (8 v_wmma per wave), then writes C via LDS as coalesced b128.
// ---------------------------------------------------------------------------
__global__ __launch_bounds__(256) void k_gemm_wmma(const u16* __restrict__ znp,
                                                   const u16* __restrict__ cnp,
                                                   float* __restrict__ dbuf) {
    __shared__ __align__(32) u16   sA[128 * ED];     // 16 KB
    __shared__ __align__(32) u16   sB[64 * ED];      //  8 KB
    __shared__ __align__(16) float sC[8][16 * 64];   // 32 KB (per-wave C tile)
    const int tid = threadIdx.x;
    const int m0 = blockIdx.y * 128;
    const int n0 = blockIdx.x * 64;

#if USE_TDM
    if (tid < 32) {   // wave 0 issues both DMA descriptors (EXEC-independent)
        unsigned ldsA = (unsigned)(uintptr_t)(void*)sA;  // flat LDS aperture: [31:0] = offset
        unsigned ldsB = (unsigned)(uintptr_t)(void*)sB;
        tdm_load_tile(ldsA, znp + (size_t)m0 * ED, 128u, (unsigned)NF);
        tdm_load_tile(ldsB, cnp + (size_t)n0 * ED, 64u,  (unsigned)NE);
        __builtin_amdgcn_s_wait_tensorcnt(0);
    }
    __builtin_prefetch(znp + (size_t)(m0 + 128) * ED, 0, 0);
    __syncthreads();
#else
    // fallback: cooperative uint4 staging (A: 1024, B: 512 vectors)
    const uint4* gA = (const uint4*)(znp + (size_t)m0 * ED);
    const uint4* gB = (const uint4*)(cnp + (size_t)n0 * ED);
    uint4* lA = (uint4*)sA;
    uint4* lB = (uint4*)sB;
    #pragma unroll
    for (int i = 0; i < 4; ++i) lA[tid + i * 256] = gA[tid + i * 256];
    #pragma unroll
    for (int i = 0; i < 2; ++i) lB[tid + i * 256] = gB[tid + i * 256];
    __builtin_prefetch(gA + 1024 + tid, 0, 0);
    __syncthreads();
#endif

    const int w    = tid >> 5;
    const int lane = tid & 31;
    const int r    = lane & 15;
    const int half = lane >> 4;

    // A fragment: lanes 0-15: M=r, K=0..15 ; lanes 16-31: M=r, K=16..31 (+32 for a1)
    const u16* aBase = sA + (w * 16 + r) * ED + half * 16;
    v16bf a0 = *(const v16bf*)(aBase);        // K =  0..31
    v16bf a1 = *(const v16bf*)(aBase + 32);   // K = 32..63

    #pragma unroll
    for (int nt = 0; nt < 4; ++nt) {
        const u16* bBase = sB + (nt * 16 + r) * ED + half * 16;
        v16bf b0 = *(const v16bf*)(bBase);
        v16bf b1 = *(const v16bf*)(bBase + 32);
        v8f c = {0.f, 0.f, 0.f, 0.f, 0.f, 0.f, 0.f, 0.f};
        c = __builtin_amdgcn_wmma_f32_16x16x32_bf16(false, a0, false, b0,
                                                    (short)0, c, false, false);
        c = __builtin_amdgcn_wmma_f32_16x16x32_bf16(false, a1, false, b1,
                                                    (short)0, c, false, false);
        // C layout: VGPR i -> M = half*8 + i, N = r ; stage into per-wave LDS
        #pragma unroll
        for (int i = 0; i < 8; ++i)
            sC[w][(half * 8 + i) * 64 + nt * 16 + r] = c[i];
    }

    // coalesced write-back: 16 rows x 64 cols per wave, float4 per lane
    #pragma unroll
    for (int it = 0; it < 8; ++it) {
        const int row  = it * 2 + (lane >> 4);
        const int col4 = lane & 15;
        float4 v = ((const float4*)&sC[w][row * 64])[col4];
        ((float4*)(dbuf + (size_t)(m0 + w * 16 + row) * NE + n0))[col4] = v;
    }
}

// ---------------------------------------------------------------------------
// 3) Row pass: per row argmax over 1024 cols; emit z_q (=z_q_st=z_output),
//    idx (as float), code-usage histogram, sq-diff loss accumulation.
// ---------------------------------------------------------------------------
__global__ __launch_bounds__(256) void k_rowpass(const float* __restrict__ dbuf,
                                                 const float* __restrict__ z,
                                                 const float* __restrict__ emb,
                                                 float* __restrict__ out,
                                                 int* __restrict__ rowidx,
                                                 int* __restrict__ ecnt,
                                                 float* __restrict__ scal) {
    const int r   = blockIdx.x;
    const int tid = threadIdx.x;
    __shared__ u64   red[256];
    __shared__ float sq[64];
    __shared__ int   sidx;

    float best = -3.0e38f; int bcol = 0;
    #pragma unroll
    for (int k = 0; k < 4; ++k) {
        int   c = tid + k * 256;
        float v = dbuf[(size_t)r * NE + c];
        if (v > best) { best = v; bcol = c; }
    }
    // ties -> lowest column (matches jnp.argmax)
    red[tid] = ((u64)fkey(best) << 32) | (u32)(NE - 1 - bcol);
    __syncthreads();
    for (int s = 128; s > 0; s >>= 1) {
        if (tid < s) { u64 o = red[tid + s]; if (o > red[tid]) red[tid] = o; }
        __syncthreads();
    }
    if (tid == 0) {
        int idx = NE - 1 - (int)(red[0] & 0xFFFFFFFFull);
        sidx = idx;
        rowidx[r] = idx;
        out[O_IDX + r] = (float)idx;
        atomicAdd(&ecnt[idx], 1);
    }
    __syncthreads();
    const int idx = sidx;

    if (tid < ED) {
        float zq = emb[(size_t)idx * ED + tid];
        float zv = z[(size_t)r * ED + tid];        // zf view of z
        float df = zq - zv;
        out[O_ZQST + (size_t)r * ED + tid] = zq;   // straight-through forward
        out[O_ZOUT + (size_t)r * ED + tid] = zq;
        sq[tid] = df * df;
    }
    __syncthreads();
    if (tid == 0) {
        float s = 0.f;
        #pragma unroll
        for (int i = 0; i < ED; ++i) s += sq[i];
        atomicAdd(&scal[0], s);
    }
}

// ---------------------------------------------------------------------------
// 4) Column max/argmax + per-column histogram over [-1,1].
//    Block: 32 columns x 4096 rows; LDS-privatized histogram (32 KB).
// ---------------------------------------------------------------------------
__global__ __launch_bounds__(256) void k_colmaxhist(const float* __restrict__ dbuf,
                                                    u64* __restrict__ colpack,
                                                    int* __restrict__ hist) {
    __shared__ int h[32 * NBINS];
    __shared__ u64 pk[32];
    const int tid  = threadIdx.x;
    const int col0 = blockIdx.x * 32;
    const int r0   = blockIdx.y * 4096;

    for (int i = tid; i < 32 * NBINS; i += 256) h[i] = 0;
    if (tid < 32) pk[tid] = 0ull;
    __syncthreads();

    const int c  = tid & 31;
    const int rs = tid >> 5;
    u64 local = 0ull;
    for (int it = 0; it < 512; ++it) {
        int   row = r0 + rs + it * 8;
        float v   = dbuf[(size_t)row * NE + col0 + c];
        int   bin = (int)((v + 1.0f) * (0.5f * NBINS));
        bin = bin < 0 ? 0 : (bin > NBINS - 1 ? NBINS - 1 : bin);
        atomicAdd(&h[c * NBINS + bin], 1);
        u64 key = ((u64)fkey(v) << 32) | (u32)(0xFFFFFFFFu - (u32)row); // ties->low row
        if (key > local) local = key;
    }
    atomicMax(&pk[c], local);
    __syncthreads();
    if (tid < 32) atomicMax(&colpack[col0 + tid], pk[tid]);
    for (int i = tid; i < 32 * NBINS; i += 256)
        atomicAdd(&hist[(size_t)col0 * NBINS + i], h[i]);
}

// ---------------------------------------------------------------------------
// 5) Per-column threshold selection from the histogram:
//    thr_top ~ 64th largest, thr_med ~ 32768th smallest (median).
// ---------------------------------------------------------------------------
__global__ __launch_bounds__(256) void k_colselect(const int* __restrict__ hist,
                                                   float* __restrict__ thrtop,
                                                   float* __restrict__ thrmed) {
    const int col = blockIdx.x * 256 + threadIdx.x;
    if (col >= NE) return;
    const int* hc = hist + (size_t)col * NBINS;
    int cnt = 0, bt = NBINS - 1;
    for (int b = NBINS - 1; b >= 0; --b) { cnt += hc[b]; if (cnt >= 64) { bt = b; break; } }
    thrtop[col] = -1.0f + (2.0f / NBINS) * (float)bt;          // lower edge
    int cum = 0, bm = NBINS - 1;
    for (int b = 0; b < NBINS; ++b) { cum += hc[b]; if (cum >= NF / 2) { bm = b; break; } }
    thrmed[col] = -1.0f + (2.0f / NBINS) * (float)(bm + 1);    // upper edge
}

// ---------------------------------------------------------------------------
// 6) Column stats: top-64 sum/count (v >= thr_top) and
//    sum exp(v/T - thr_med/T) over the bottom half (v <= thr_med).
// ---------------------------------------------------------------------------
__global__ __launch_bounds__(256) void k_colstats(const float* __restrict__ dbuf,
                                                  const float* __restrict__ thrtop,
                                                  const float* __restrict__ thrmed,
                                                  float* __restrict__ topsum,
                                                  int* __restrict__ topcnt,
                                                  float* __restrict__ sumexp) {
    __shared__ float tsum[32];
    __shared__ int   tcnt[32];
    __shared__ float sexp[32];
    const int tid  = threadIdx.x;
    const int col0 = blockIdx.x * 32;
    const int r0   = blockIdx.y * 4096;
    if (tid < 32) { tsum[tid] = 0.f; tcnt[tid] = 0; sexp[tid] = 0.f; }
    __syncthreads();

    const int c  = tid & 31;
    const int rs = tid >> 5;
    const float tT = thrtop[col0 + c];
    const float tM = thrmed[col0 + c];
    const float mm = tM * INV_T;
    float ls = 0.f, le = 0.f; int lc = 0;
    for (int it = 0; it < 512; ++it) {
        int   row = r0 + rs + it * 8;
        float v   = dbuf[(size_t)row * NE + col0 + c];
        if (v >= tT) { ls += v; lc += 1; }
        if (v <= tM) { le += __expf(v * INV_T - mm); }
    }
    atomicAdd(&tsum[c], ls);
    atomicAdd(&tcnt[c], lc);
    atomicAdd(&sexp[c], le);
    __syncthreads();
    if (tid < 32) {
        atomicAdd(&topsum[col0 + tid], tsum[tid]);
        atomicAdd(&topcnt[col0 + tid], tcnt[tid]);
        atomicAdd(&sumexp[col0 + tid], sexp[tid]);
    }
}

// ---------------------------------------------------------------------------
// 7) Per-embedding finalize: new_embed_prob, new_emb (anchor gather),
//    contra and entropy partials. One 64-thread block per embedding.
// ---------------------------------------------------------------------------
__global__ __launch_bounds__(64) void k_fincols(const float* __restrict__ z,
                                                const float* __restrict__ emb,
                                                const float* __restrict__ eprob,
                                                const u64* __restrict__ colpack,
                                                const int* __restrict__ ecnt,
                                                const float* __restrict__ topsum,
                                                const int* __restrict__ topcnt,
                                                const float* __restrict__ thrmed,
                                                const float* __restrict__ sumexp,
                                                float* __restrict__ out,
                                                float* __restrict__ scal) {
    const int e = blockIdx.x;
    const int t = threadIdx.x;
    const u32 anchor = 0xFFFFFFFFu - (u32)(colpack[e] & 0xFFFFFFFFull);
    const float em   = (float)ecnt[e] * (1.0f / (float)NF);
    const float nep  = eprob[e] * 0.99f + em * 0.01f;
    const float dec  = __expf(-(nep * (float)NE * 10.0f) * 100.0f - 0.001f); // /(1-DECAY)
    if (t == 0) out[O_NEP + e] = nep;
    const float rf = z[(size_t)anchor * ED + t];
    out[O_NEMB + (size_t)e * ED + t] = emb[(size_t)e * ED + t] * (1.0f - dec) + rf * dec;
    if (t == 0) {
        float dp  = (topsum[e] / fmaxf((float)topcnt[e], 1.0f)) * INV_T;
        float m   = thrmed[e] * INV_T;
        float lse = __logf(__expf(dp - m) + sumexp[e]) + m;
        atomicAdd(&scal[1], (lse - dp) * (1.0f / (float)NE));
        atomicAdd(&scal[2], em * __logf(em + 1e-10f));
    }
}

// ---------------------------------------------------------------------------
// 8) One-hot rewrite of the d buffer -> min_encodings (must run last among
//    consumers of d). float4 stores, 4 cols/thread.
// ---------------------------------------------------------------------------
__global__ __launch_bounds__(256) void k_onehot(const int* __restrict__ rowidx,
                                                float* __restrict__ out) {
    const int r   = blockIdx.x;
    const int idx = rowidx[r];
    const int c0  = threadIdx.x * 4;
    float4 v;
    v.x = (c0 + 0 == idx) ? 1.0f : 0.0f;
    v.y = (c0 + 1 == idx) ? 1.0f : 0.0f;
    v.z = (c0 + 2 == idx) ? 1.0f : 0.0f;
    v.w = (c0 + 3 == idx) ? 1.0f : 0.0f;
    ((float4*)(out + O_ME + (size_t)r * NE))[threadIdx.x] = v;
}

// ---------------------------------------------------------------------------
// 9) Scalar outputs.
// ---------------------------------------------------------------------------
__global__ void k_scalars(const float* __restrict__ scal, float* __restrict__ out) {
    float l1 = scal[0] * (1.0f / 4194304.0f);   // mean over B*NSUB*ED
    out[O_LOSS1] = l1;
    out[O_LOSS2] = l1;                           // forward-equal
    out[O_LOSS]  = 1.0f * l1 + 0.25f * l1 + scal[1];
    out[O_PERP]  = __expf(-scal[2]);
}

// ---------------------------------------------------------------------------
// Launch
// ---------------------------------------------------------------------------
extern "C" void kernel_launch(void* const* d_in, const int* in_sizes, int n_in,
                              void* d_out, int out_size, void* d_ws, size_t ws_size,
                              hipStream_t stream) {
    const float* z     = (const float*)d_in[0];
    const float* emb   = (const float*)d_in[1];
    const float* eprob = (const float*)d_in[2];
    float* out = (float*)d_out;
    char*  ws  = (char*)d_ws;

    u16*   zn      = (u16*)(ws + WS_ZN);
    u16*   cn      = (u16*)(ws + WS_CN);
    int*   rowidx  = (int*)(ws + WS_ROWIDX);
    u64*   colpack = (u64*)(ws + WS_COLPACK);
    int*   hist    = (int*)(ws + WS_HIST);
    float* thrtop  = (float*)(ws + WS_THRTOP);
    float* thrmed  = (float*)(ws + WS_THRMED);
    float* topsum  = (float*)(ws + WS_TOPSUM);
    int*   topcnt  = (int*)(ws + WS_TOPCNT);
    float* sumexp  = (float*)(ws + WS_SUMEXP);
    int*   ecnt    = (int*)(ws + WS_ECNT);
    float* scal    = (float*)(ws + WS_SCAL);
    float* dbuf    = out + O_ME;   // 268 MB staging == min_encodings slot

    // zero accumulators / histograms (graph-capture-safe)
    hipMemsetAsync(ws + WS_COLPACK, 0, WS_END - WS_COLPACK, stream);

    k_norm_rows<<<NF / 8, 256, 0, stream>>>(z, zn, NF);
    k_norm_rows<<<NE / 8, 256, 0, stream>>>(emb, cn, NE);
    k_gemm_wmma<<<dim3(NE / 64, NF / 128), 256, 0, stream>>>(zn, cn, dbuf);
    k_rowpass<<<NF, 256, 0, stream>>>(dbuf, z, emb, out, rowidx, ecnt, scal);
    k_colmaxhist<<<dim3(NE / 32, NF / 4096), 256, 0, stream>>>(dbuf, colpack, hist);
    k_colselect<<<NE / 256, 256, 0, stream>>>(hist, thrtop, thrmed);
    k_colstats<<<dim3(NE / 32, NF / 4096), 256, 0, stream>>>(dbuf, thrtop, thrmed,
                                                             topsum, topcnt, sumexp);
    k_fincols<<<NE, ED, 0, stream>>>(z, emb, eprob, colpack, ecnt, topsum, topcnt,
                                     thrmed, sumexp, out, scal);
    k_onehot<<<NF, 256, 0, stream>>>(rowidx, out);
    k_scalars<<<1, 1, 0, stream>>>(scal, out);
}
